// Seq2SeqForecast_89378269430395
// MI455X (gfx1250) — compile-verified, matrix operand order checked by reading, not run on hardware
//
#include <hip/hip_runtime.h>
#include <hip/hip_bf16.h>

// ---------------------------------------------------------------------------
// Seq2Seq GRU forecaster for MI455X (gfx1250), wave32 + WMMA bf16.
// Batch-partitioned persistent kernel: 32 WGs x 256 threads, each WG owns 16
// batch rows and runs encoder(2 fused GRU layers, T=336) -> bridge linear ->
// decoder GRU (48 steps) -> fc head, all without grid sync. Weights are
// pre-packed to bf16 WMMA B-fragments (L2-resident, ~7MB).
//
// Register strategy (rounds 3-6 spilled): each wave's 4 hidden tiles are
// processed as TWO sequential half-groups of 2. Per half: 64 acc VGPRs +
// 3 rotating 2-frag B buffers (48 VGPRs) + 16-reg held h_new. Peak live
// ~140 VGPRs -> large headroom under the ~512 cap (256 thr = 2 waves/SIMD),
// so the scheduler's load hoisting becomes free pipelining instead of spills.
// Gates read h_old from an f32 LDS mirror (reads are race-free pre-barrier);
// h_new stores happen after the barrier, preserving RAW discipline.
// ---------------------------------------------------------------------------

typedef __attribute__((ext_vector_type(16))) __bf16 v16bf;
typedef __attribute__((ext_vector_type(8)))  __bf16 v8bf;
typedef __attribute__((ext_vector_type(8)))  float  v8f;

#define HDIM   512
#define TDIM   336
#define HORIZ  48
#define BATCH  512
#define TB     16          // batch rows per workgroup (one WMMA M tile)
#define NWAVES 8
#define JPW    4           // hidden tiles per wave: (512/16)/8
#define GW     2           // tiles per half-group
#define KB_H   16          // K tiles for K=512 (K step = 32)

static __device__ __forceinline__ v8f splat8(float x) {
  v8f v;
#pragma unroll
  for (int i = 0; i < 8; i++) v[i] = x;
  return v;
}

static __device__ __forceinline__ float sigmoid_f(float x) {
  return 1.f / (1.f + __expf(-x));
}
static __device__ __forceinline__ float tanh_f(float x) {
  float e = __expf(-2.f * fabsf(x));
  float t = (1.f - e) / (1.f + e);
  return x < 0.f ? -t : t;
}

// A fragment (16x32 bf16) from row-major LDS buffer.
// Lane layout (ISA 7.12.2): lanes 0-15: M=lane, K={k0..k0+7, k0+16..k0+23};
// lanes 16-31: M=lane-16, K={k0+8..k0+15, k0+24..k0+31}.
static __device__ __forceinline__ v16bf lds_afrag(const __bf16* buf, int rowstride, int k0) {
  const int lane = threadIdx.x & 31;
  const int m    = lane & 15;
  const int off  = (lane & 16) ? 8 : 0;
  const __bf16* p = buf + m * rowstride + k0 + off;
  v8bf lo = *(const v8bf*)(p);
  v8bf hi = *(const v8bf*)(p + 16);
  v16bf a;
#pragma unroll
  for (int i = 0; i < 8; i++) { a[i] = lo[i]; a[i + 8] = hi[i]; }
  return a;
}

// B fragment (32x16 bf16) from pre-packed global weights: lane loads its
// 16 contiguous bf16 (32 bytes -> two b128 loads, L2-resident).
static __device__ __forceinline__ v16bf g_bfrag(const __bf16* pw, int frag) {
  const int lane = threadIdx.x & 31;
  return *(const v16bf*)(pw + ((size_t)frag * 32 + lane) * 16);
}

static __device__ __forceinline__ v8f wmma_bf16(v16bf a, v16bf b, v8f c) {
  return __builtin_amdgcn_wmma_f32_16x16x32_bf16(false, a, false, b, (short)0, c, false, false);
}

// Load GW B fragments (one gate group) for K stage kb.
static __device__ __forceinline__ void load_bg(v16bf (&b)[GW], const __bf16* pW,
                                               int gt0, int KB, int kb) {
#pragma unroll
  for (int j = 0; j < GW; j++) b[j] = g_bfrag(pW, (gt0 + j) * KB + kb);
}

// GW independent WMMAs sharing one A fragment.
static __device__ __forceinline__ void wmmag(v16bf a, v16bf (&b)[GW], v8f (&acc)[GW]) {
#pragma unroll
  for (int j = 0; j < GW; j++) acc[j] = wmma_bf16(a, b[j], acc[j]);
}

// Triple-output GEMM over K, software-pipelined with 3 rotating GW-frag B
// buffers; each load is issued 2 sub-stages ahead of its use.
// Works for any KB >= 1 (KB==1 runs only the peeled tail).
template <int KB>
static __device__ __forceinline__ void gemm3g(
    const __bf16* abuf, int astride, const __bf16* pW, int jt0g,
    v8f (&ar)[GW], v8f (&az)[GW], v8f (&a3)[GW]) {
  v16bf B0[GW], B1[GW], B2[GW];
  load_bg(B0, pW, jt0g, KB, 0);            // (kb=0, gate r)
  load_bg(B1, pW, jt0g + 32, KB, 0);       // (kb=0, gate z)
#pragma unroll 1
  for (int kb = 0; kb < KB - 1; kb++) {
    v16bf a = lds_afrag(abuf, astride, kb * 32);
    load_bg(B2, pW, jt0g + 64, KB, kb);    // (kb, gate n)
    wmmag(a, B0, ar);
    load_bg(B0, pW, jt0g, KB, kb + 1);     // (kb+1, gate r)
    wmmag(a, B1, az);
    load_bg(B1, pW, jt0g + 32, KB, kb + 1);// (kb+1, gate z)
    wmmag(a, B2, a3);
  }
  // tail: kb == KB-1 (B0,B1 already loaded)
  v16bf a = lds_afrag(abuf, astride, (KB - 1) * 32);
  load_bg(B2, pW, jt0g + 64, KB, KB - 1);
  wmmag(a, B0, ar);
  wmmag(a, B1, az);
  wmmag(a, B2, a3);
}

// One half-group (GW hidden tiles) of a GRU step: GEMMs + gates.
// h_new is returned in registers; nothing is written to LDS here.
template <int KBIN>
static __device__ __forceinline__ void gru_half(
    const __bf16* inbuf, int instride, const __bf16* hbuf,
    const __bf16* pWih, const __bf16* pWhh,
    const float* bih, const float* bhh, const float* hf32,
    int g, v8f (&hnew)[GW]) {
  const int lane  = threadIdx.x & 31;
  const int wave  = threadIdx.x >> 5;
  const int n     = lane & 15;
  const int jt0g  = wave * JPW + g * GW;
  v8f ar[GW], az[GW], anh[GW], ani[GW];
#pragma unroll
  for (int j = 0; j < GW; j++) {
    const int jt = jt0g + j;
    ar[j]  = splat8(bih[jt * 16 + n] + bhh[jt * 16 + n]);
    az[j]  = splat8(bih[(jt + 32) * 16 + n] + bhh[(jt + 32) * 16 + n]);
    anh[j] = splat8(bhh[(jt + 64) * 16 + n]);
    ani[j] = splat8(bih[(jt + 64) * 16 + n]);
  }
  gemm3g<KBIN>(inbuf, instride, pWih, jt0g, ar, az, ani);  // input projection
  gemm3g<KB_H>(hbuf, HDIM, pWhh, jt0g, ar, az, anh);       // recurrent proj.
  // Gates on fragment registers; h_old read from f32 LDS mirror (read-only).
  // D layout: VGPR e -> M = e + (lane>=16 ? 8 : 0), N = lane&15.
  const int mbase = (lane & 16) ? 8 : 0;
  const int ncol  = lane & 15;
#pragma unroll
  for (int j = 0; j < GW; j++) {
    const int col = (jt0g + j) * 16 + ncol;
#pragma unroll
    for (int e = 0; e < 8; e++) {
      float hold = hf32[(mbase + e) * HDIM + col];
      float r  = sigmoid_f(ar[j][e]);
      float z  = sigmoid_f(az[j][e]);
      float nn = tanh_f(ani[j][e] + r * anh[j][e]);
      hnew[j][e] = (1.f - z) * nn + z * hold;
    }
  }
}

// Store one half-group's h_new to both the f32 mirror and bf16 buffer.
static __device__ __forceinline__ void store_half(v8f (&hf)[GW], int g,
                                                  float* hf32, __bf16* hbf) {
  const int lane  = threadIdx.x & 31;
  const int wave  = threadIdx.x >> 5;
  const int mbase = (lane & 16) ? 8 : 0;
  const int ncol  = lane & 15;
  const int jt0g  = wave * JPW + g * GW;
#pragma unroll
  for (int j = 0; j < GW; j++) {
    const int col = (jt0g + j) * 16 + ncol;
#pragma unroll
    for (int e = 0; e < 8; e++) {
      const int idx = (mbase + e) * HDIM + col;
      hf32[idx] = hf[j][e];
      hbf[idx]  = (__bf16)hf[j][e];
    }
  }
}

// Full GRU layer step: two half-groups, then barrier-protected state update.
template <int KBIN>
static __device__ __forceinline__ void gru_layer(
    const __bf16* inbuf, int instride, __bf16* hbf, float* hf32,
    const __bf16* pWih, const __bf16* pWhh,
    const float* bih, const float* bhh) {
  v8f hn0[GW], hn1[GW];
  gru_half<KBIN>(inbuf, instride, hbf, pWih, pWhh, bih, bhh, hf32, 0, hn0);
  gru_half<KBIN>(inbuf, instride, hbf, pWih, pWhh, bih, bhh, hf32, 1, hn1);
  __syncthreads();                 // all waves done reading h_old / inputs
  store_half(hn0, 0, hf32, hbf);
  store_half(hn1, 1, hf32, hbf);
  __syncthreads();                 // new state visible to all waves
}

// ---------------------------------------------------------------------------
// Weight packing: f32 (G x Kin) row-major -> bf16 WMMA B-fragments.
// frag = g_tile*KB + kb; element (frag, lane, i): value = W[g_tile*16 + lane%16,
// kb*32 + kmap(lane,i)], zero-padded past Kin. kmap matches the A interleave.
// ---------------------------------------------------------------------------
__global__ void pack_w(const float* __restrict__ W, __bf16* __restrict__ out,
                       int Kin, int KB, int GT) {
  int idx = blockIdx.x * blockDim.x + threadIdx.x;
  int total = GT * KB * 512;
  if (idx >= total) return;
  int i    = idx & 15;
  int lane = (idx >> 4) & 31;
  int frag = idx >> 9;
  int kb   = frag % KB;
  int gt   = frag / KB;
  int n    = lane & 15;
  int k    = (i < 8 ? i : i + 8) + ((lane & 16) ? 8 : 0);
  int gk   = kb * 32 + k;
  float v  = (gk < Kin) ? W[(size_t)(gt * 16 + n) * Kin + gk] : 0.f;
  out[idx] = (__bf16)v;
}

// ---------------------------------------------------------------------------
// Main persistent kernel: one WG per 16 batch rows.
// ---------------------------------------------------------------------------
__global__ __launch_bounds__(256, 1) void seq2seq_kernel(
    const float* __restrict__ src, const float* __restrict__ tf,
    const float* __restrict__ bih0, const float* __restrict__ bhh0,
    const float* __restrict__ bih1, const float* __restrict__ bhh1,
    const float* __restrict__ dec0b,
    const float* __restrict__ cbih, const float* __restrict__ cbhh,
    const float* __restrict__ fcW, const float* __restrict__ fcb,
    const __bf16* __restrict__ pWih0, const __bf16* __restrict__ pWhh0,
    const __bf16* __restrict__ pWih1, const __bf16* __restrict__ pWhh1,
    const __bf16* __restrict__ pDec0, const __bf16* __restrict__ pCWih,
    const __bf16* __restrict__ pCWhh, float* __restrict__ out) {
  __shared__ __bf16 hbf1[TB * HDIM];   // layer-0 h / decoder hidden (bf16)
  __shared__ __bf16 hbf2[TB * HDIM];   // layer-1 h (bf16)
  __shared__ float  hf32_1[TB * HDIM]; // f32 mirrors of the hidden state
  __shared__ float  hf32_2[TB * HDIM];
  __shared__ __bf16 xbuf[TB * 32];     // zero-padded K=32 step input
  __shared__ float  predbuf[TB];
  __shared__ float  red[TB * 16];      // fc-head reduction

  const int b0  = blockIdx.x * TB;
  const int tid = threadIdx.x;

  for (int i = tid; i < TB * HDIM; i += 256) {
    hbf1[i] = (__bf16)0.f; hbf2[i] = (__bf16)0.f;
    hf32_1[i] = 0.f; hf32_2[i] = 0.f;
  }
  for (int i = tid; i < TB * 32; i += 256) xbuf[i] = (__bf16)0.f;
  __syncthreads();

  // -------- Encoder: two fused GRU layers over T=336 --------
  for (int t = 0; t < TDIM; t++) {
    if (tid < TB) {
      const float* sp = src + ((size_t)(b0 + tid) * TDIM + t) * 8;
#pragma unroll
      for (int c = 0; c < 8; c++) xbuf[tid * 32 + c] = (__bf16)sp[c];
    }
    __syncthreads();
    gru_layer<1>(xbuf, 32, hbf1, hf32_1, pWih0, pWhh0, bih0, bhh0);
    gru_layer<KB_H>(hbf1, HDIM, hbf2, hf32_2, pWih1, pWhh1, bih1, bhh1);
  }

  // -------- Bridge: hidden = h2_last @ dec0_W^T + dec0_b (into hbf1) ------
  {
    const int lane = tid & 31;
    const int wave = tid >> 5;
    const int n    = lane & 15;
#pragma unroll
    for (int g = 0; g < 2; g++) {
      const int jt0g = wave * JPW + g * GW;
      v8f acc[GW];
#pragma unroll
      for (int j = 0; j < GW; j++) acc[j] = splat8(dec0b[(jt0g + j) * 16 + n]);
      v16bf C0[GW], C1[GW];
      load_bg(C0, pDec0, jt0g, KB_H, 0);
#pragma unroll 1
      for (int kb = 0; kb < KB_H - 1; kb += 2) {
        load_bg(C1, pDec0, jt0g, KB_H, kb + 1);
        v16bf aE = lds_afrag(hbf2, HDIM, kb * 32);
        wmmag(aE, C0, acc);
        load_bg(C0, pDec0, jt0g, KB_H, kb + 2 < KB_H ? kb + 2 : kb + 1);
        v16bf aO = lds_afrag(hbf2, HDIM, (kb + 1) * 32);
        wmmag(aO, C1, acc);
      }
      store_half(acc, g, hf32_1, hbf1);   // no reader of hbf1 before barrier
    }
  }
  if (tid < TB)
    predbuf[tid] = src[((size_t)(b0 + tid) * TDIM + (TDIM - 1)) * 8 + 0];
  __syncthreads();

  // -------- Decoder: 48 autoregressive GRU-cell steps + fc head -----------
  for (int t = 0; t < HORIZ; t++) {
    if (tid < TB) {
      xbuf[tid * 32 + 0] = (__bf16)predbuf[tid];
      const float* tp = tf + ((size_t)(b0 + tid) * HORIZ + t) * 8;
#pragma unroll
      for (int c = 0; c < 8; c++) xbuf[tid * 32 + 1 + c] = (__bf16)tp[c];
    }
    __syncthreads();
    gru_layer<1>(xbuf, 32, hbf1, hf32_1, pCWih, pCWhh, cbih, cbhh);
    // fc head: 16 threads/row, 32-wide segments, LDS tree reduce.
    {
      const int row = tid >> 4, seg = tid & 15;
      float s = 0.f;
      const int kb = seg * 32;
      for (int k = kb; k < kb + 32; k++) s += (float)hbf1[row * HDIM + k] * fcW[k];
      red[row * 16 + seg] = s;
    }
    __syncthreads();
    if (tid < TB) {
      float s = fcb[0];
#pragma unroll
      for (int q = 0; q < 16; q++) s += red[tid * 16 + q];
      out[(size_t)(b0 + tid) * HORIZ + t] = s;
      predbuf[tid] = s;
    }
    __syncthreads();
  }
}

// ---------------------------------------------------------------------------
extern "C" void kernel_launch(void* const* d_in, const int* in_sizes, int n_in,
                              void* d_out, int out_size, void* d_ws, size_t ws_size,
                              hipStream_t stream) {
  const float* src   = (const float*)d_in[0];
  const float* tf    = (const float*)d_in[1];
  const float* Wih0  = (const float*)d_in[2];
  const float* Whh0  = (const float*)d_in[3];
  const float* bih0  = (const float*)d_in[4];
  const float* bhh0  = (const float*)d_in[5];
  const float* Wih1  = (const float*)d_in[6];
  const float* Whh1  = (const float*)d_in[7];
  const float* bih1  = (const float*)d_in[8];
  const float* bhh1  = (const float*)d_in[9];
  const float* dec0W = (const float*)d_in[10];
  const float* dec0b = (const float*)d_in[11];
  const float* cWih  = (const float*)d_in[12];
  const float* cWhh  = (const float*)d_in[13];
  const float* cbih  = (const float*)d_in[14];
  const float* cbhh  = (const float*)d_in[15];
  const float* fcW   = (const float*)d_in[16];
  const float* fcb   = (const float*)d_in[17];
  float* out = (float*)d_out;

  // Workspace: bf16-packed B-fragment weights (~7 MB total).
  __bf16* ws = (__bf16*)d_ws;
  size_t o = 0;
  __bf16* pWih0 = ws + o; o += (size_t)96 * 1 * 512;
  __bf16* pWhh0 = ws + o; o += (size_t)96 * 16 * 512;
  __bf16* pWih1 = ws + o; o += (size_t)96 * 16 * 512;
  __bf16* pWhh1 = ws + o; o += (size_t)96 * 16 * 512;
  __bf16* pDec0 = ws + o; o += (size_t)32 * 16 * 512;
  __bf16* pCWih = ws + o; o += (size_t)96 * 1 * 512;
  __bf16* pCWhh = ws + o; o += (size_t)96 * 16 * 512;

  auto pack = [&](const float* W, __bf16* dst, int Kin, int KB, int GT) {
    int total = GT * KB * 512;
    pack_w<<<(total + 255) / 256, 256, 0, stream>>>(W, dst, Kin, KB, GT);
  };
  pack(Wih0, pWih0, 8, 1, 96);
  pack(Whh0, pWhh0, 512, 16, 96);
  pack(Wih1, pWih1, 512, 16, 96);
  pack(Whh1, pWhh1, 512, 16, 96);
  pack(dec0W, pDec0, 512, 16, 32);
  pack(cWih, pCWih, 9, 1, 96);
  pack(cWhh, pCWhh, 512, 16, 96);

  seq2seq_kernel<<<BATCH / TB, 256, 0, stream>>>(
      src, tf, bih0, bhh0, bih1, bhh1, dec0b, cbih, cbhh, fcW, fcb,
      pWih0, pWhh0, pWih1, pWhh1, pDec0, pCWih, pCWhh, out);
}
